// MultiHeadAttention_9904194584645
// MI455X (gfx1250) — matrix operand order repaired and checked by reading, not compiled, not run on hardware
//
#include <hip/hip_runtime.h>

#define D_MODEL 1024
#define SEQ     2048
#define NB      2
#define NH      16
#define DH      64
#define MTOT    (NB * SEQ) /* 4096 */

typedef __attribute__((ext_vector_type(16))) __bf16 v16bf;
typedef __attribute__((ext_vector_type(8)))  __bf16 v8bf;
typedef __attribute__((ext_vector_type(8)))  float  v8f;

__device__ __forceinline__ unsigned short f2bf(float f) {
  unsigned int u = __float_as_uint(f);
  u += 0x7FFFu + ((u >> 16) & 1u);           // round-to-nearest-even
  return (unsigned short)(u >> 16);
}

__device__ __forceinline__ v8bf ld8(const unsigned short* p) {
  uint4 u = *reinterpret_cast<const uint4*>(p);  // global_load_b128 / ds_load_b128
  return __builtin_bit_cast(v8bf, u);
}

__device__ __forceinline__ v16bf frag(v8bf lo, v8bf hi) {
  return __builtin_shufflevector(lo, hi, 0,1,2,3,4,5,6,7,8,9,10,11,12,13,14,15);
}

__device__ __forceinline__ v8f wmma_bf16(v16bf a, v16bf b, v8f c) {
  return __builtin_amdgcn_wmma_f32_16x16x32_bf16(
      /*neg_a=*/false, a, /*neg_b=*/false, b,
      /*c_mod=*/(short)0, c, /*reuse_a=*/false, /*reuse_b=*/false);
}

// ---------------------------------------------------------------- convert
__global__ void cvt_f32_bf16(const float* __restrict__ src,
                             unsigned short* __restrict__ dst, int n) {
  int i = blockIdx.x * blockDim.x + threadIdx.x;
  if (i < n) dst[i] = f2bf(src[i]);
}

// ---------------------------------------------------------------- GEMM
// C[M,N] = A[M,K](bf16) @ W[N,K]^T(bf16) + bias.
// Block = 128 threads (4 waves); block tile = 128x64; wave tile = 32x64.
// W tile (64 cols x 32 k = 4KB) streamed into LDS with double-buffered
// GLOBAL_LOAD_ASYNC_TO_LDS_B128 (ASYNCcnt), shared by all 4 waves.
// mode 0: store bf16 [b,h,s,d] (Q/K); mode 1: store bf16 [b,h,d,s] (V^T);
// mode 2: store f32 row-major [M,N] (final output).
__global__ void __launch_bounds__(128)
gemm_wmma(const unsigned short* __restrict__ A,
          const unsigned short* __restrict__ W,
          const float* __restrict__ bias,
          void* __restrict__ dst,
          int M, int N, int K, float scale, int mode) {
  __shared__ unsigned short wtile[2][64][32];   // [buf][col][k] : 8KB
  const int lane  = threadIdx.x & 31;
  const int wave  = threadIdx.x >> 5;
  const int l15   = lane & 15;
  const int lh    = lane >> 4;
  const int mrow0 = blockIdx.x * 128 + wave * 32;
  const int nbase = blockIdx.y * 64;
  const unsigned ldsbase = (unsigned)(unsigned long long)&wtile[0][0][0];

  // async-stage issue: 256 x b128 tasks, 2 per thread
  const int tid   = threadIdx.x;
  auto issue_stage = [&](int buf, int k0) {
#pragma unroll
    for (int p = 0; p < 2; ++p) {
      const int id    = tid + p * 128;
      const int c     = id >> 2;          // column 0..63
      const int chunk = id & 3;           // 8-element k chunk
      unsigned lds_off = ldsbase + ((unsigned)(buf * 64 + c) * 32 + chunk * 8) * 2;
      unsigned gv_off  = ((unsigned)(nbase + c) * K + k0 + chunk * 8) * 2;
      asm volatile("global_load_async_to_lds_b128 %0, %1, %2"
                   :: "v"(lds_off), "v"(gv_off), "s"(W) : "memory");
    }
  };

  v8f acc[2][4] = {{{}, {}, {}, {}}, {{}, {}, {}, {}}};

  const unsigned short* arow0 = A + (size_t)(mrow0 + l15) * K;
  const unsigned short* arow1 = A + (size_t)(mrow0 + 16 + l15) * K;

  issue_stage(0, 0);                      // prologue
  for (int k0 = 0; k0 < K; k0 += 32) {
    const int buf = (k0 >> 5) & 1;
    if (k0 + 32 < K) {
      issue_stage(buf ^ 1, k0 + 32);
      asm volatile("s_wait_asynccnt 0x2" ::: "memory");  // current stage done
    } else {
      asm volatile("s_wait_asynccnt 0x0" ::: "memory");
    }
    __syncthreads();                      // all waves' async writes visible

    const int ka = k0 + lh * 8;           // A-frag: K interleave by lane-half
    __builtin_prefetch(arow0 + ka + 64, 0, 1);  // global_prefetch_b8
    v16bf af0 = frag(ld8(arow0 + ka), ld8(arow0 + ka + 16));
    v16bf af1 = frag(ld8(arow1 + ka), ld8(arow1 + ka + 16));
#pragma unroll
    for (int j = 0; j < 4; ++j) {         // B-frag from LDS: [col][k]
      const unsigned short* wl = &wtile[buf][j * 16 + l15][lh * 8];
      v16bf bfr = frag(ld8(wl), ld8(wl + 16));
      acc[0][j] = wmma_bf16(af0, bfr, acc[0][j]);
      acc[1][j] = wmma_bf16(af1, bfr, acc[1][j]);
    }
    __syncthreads();                      // reads done before buf reissued
  }

#pragma unroll
  for (int t = 0; t < 2; ++t)
#pragma unroll
    for (int j = 0; j < 4; ++j) {
      const int col = nbase + j * 16 + l15;
      const float bval = bias[col];
#pragma unroll
      for (int r = 0; r < 8; ++r) {
        const int row = mrow0 + t * 16 + r + lh * 8;  // lanes 16-31 hold M+8
        const float v = (acc[t][j][r] + bval) * scale;
        if (mode == 0) {
          const int b = row >> 11, s = row & (SEQ - 1);
          const int h = col >> 6,  d = col & (DH - 1);
          ((unsigned short*)dst)[(((size_t)(b * NH + h) * SEQ) + s) * DH + d] = f2bf(v);
        } else if (mode == 1) {
          const int b = row >> 11, s = row & (SEQ - 1);
          const int h = col >> 6,  d = col & (DH - 1);
          ((unsigned short*)dst)[(((size_t)(b * NH + h) * DH) + d) * SEQ + s] = f2bf(v);
        } else {
          ((float*)dst)[(size_t)row * N + col] = v;
        }
      }
    }
}

// ---------------------------------------------------------------- flash attention
// One wave owns a 16-query tile; streams causal key tiles of 32.
__global__ void flash_attn(const unsigned short* __restrict__ q,
                           const unsigned short* __restrict__ kbuf,
                           const unsigned short* __restrict__ vt,
                           unsigned short* __restrict__ o) {
  __shared__ unsigned short pbuf[4][16][48];   // per-wave P staging (16B-aligned rows)
  const int lane = threadIdx.x & 31;
  const int wave = threadIdx.x >> 5;
  const int l15  = lane & 15;
  const int lh   = lane >> 4;
  const int bh   = blockIdx.x;                          // b*NH + h
  const int q0   = (blockIdx.y * 4 + wave) * 16;

  // Q A-fragments over the two Dh halves (d 0..31 / 32..63)
  const unsigned short* qrow = q + ((size_t)bh * SEQ + q0 + l15) * DH;
  v16bf aq0 = frag(ld8(qrow + lh * 8),      ld8(qrow + lh * 8 + 16));
  v16bf aq1 = frag(ld8(qrow + 32 + lh * 8), ld8(qrow + 32 + lh * 8 + 16));

  v8f oacc[4] = {{}, {}, {}, {}};
  float mrow[8], lrow[8];
#pragma unroll
  for (int r = 0; r < 8; ++r) { mrow[r] = -1e30f; lrow[r] = 0.f; }

  for (int kt = 0; kt < q0 + 16; kt += 32) {
    // ---- S = Q @ K^T for 16x32 keys (two 16x16 C tiles)
    v8f sc[2];
#pragma unroll
    for (int jj = 0; jj < 2; ++jj) {
      const unsigned short* krow =
          kbuf + ((size_t)bh * SEQ + kt + jj * 16 + l15) * DH;
      v16bf b0 = frag(ld8(krow + lh * 8),      ld8(krow + lh * 8 + 16));
      v16bf b1 = frag(ld8(krow + 32 + lh * 8), ld8(krow + 32 + lh * 8 + 16));
      v8f s0 = {};
      s0 = wmma_bf16(aq0, b0, s0);
      s0 = wmma_bf16(aq1, b1, s0);
      sc[jj] = s0;
    }
    // ---- causal mask (elementwise in C layout)
#pragma unroll
    for (int jj = 0; jj < 2; ++jj) {
      const int key = kt + jj * 16 + l15;
#pragma unroll
      for (int r = 0; r < 8; ++r) {
        const int qq = q0 + r + lh * 8;
        if (key > qq) sc[jj][r] = -1e30f;
      }
    }
    // ---- online softmax: row reductions across each 16-lane half (wave32)
#pragma unroll
    for (int r = 0; r < 8; ++r) {
      float tm = fmaxf(sc[0][r], sc[1][r]);
      tm = fmaxf(tm, __shfl_xor(tm, 1, 32));
      tm = fmaxf(tm, __shfl_xor(tm, 2, 32));
      tm = fmaxf(tm, __shfl_xor(tm, 4, 32));
      tm = fmaxf(tm, __shfl_xor(tm, 8, 32));
      const float mnew  = fmaxf(mrow[r], tm);
      const float corr  = __expf(mrow[r] - mnew);
      mrow[r] = mnew;
      const float p0 = __expf(sc[0][r] - mnew);
      const float p1 = __expf(sc[1][r] - mnew);
      sc[0][r] = p0; sc[1][r] = p1;
      float rs = p0 + p1;
      rs += __shfl_xor(rs, 1, 32);
      rs += __shfl_xor(rs, 2, 32);
      rs += __shfl_xor(rs, 4, 32);
      rs += __shfl_xor(rs, 8, 32);
      lrow[r] = lrow[r] * corr + rs;
#pragma unroll
      for (int j = 0; j < 4; ++j) oacc[j][r] *= corr;
    }
    // ---- transpose P: C-layout -> LDS -> A-fragment layout (wave-local, in-order DS)
#pragma unroll
    for (int jj = 0; jj < 2; ++jj)
#pragma unroll
      for (int r = 0; r < 8; ++r)
        pbuf[wave][r + lh * 8][jj * 16 + l15] = f2bf(sc[jj][r]);
    const unsigned short* prow = &pbuf[wave][l15][0];
    v16bf pf = frag(ld8(prow + lh * 8), ld8(prow + lh * 8 + 16));
    // ---- O += P @ V  (V^T layout makes B-fragments contiguous in key dim)
#pragma unroll
    for (int jd = 0; jd < 4; ++jd) {
      const unsigned short* vrow =
          vt + ((size_t)bh * DH + jd * 16 + l15) * SEQ + kt + lh * 8;
      v16bf bv = frag(ld8(vrow), ld8(vrow + 16));
      oacc[jd] = wmma_bf16(pf, bv, oacc[jd]);
    }
  }

  // ---- normalize & store attn output as bf16 [b, s, h*DH+d]
  const int b = bh >> 4, h = bh & 15;
#pragma unroll
  for (int jd = 0; jd < 4; ++jd)
#pragma unroll
    for (int r = 0; r < 8; ++r) {
      const int s = q0 + r + lh * 8;
      const int e = h * DH + jd * 16 + l15;
      o[((size_t)(b * SEQ) + s) * D_MODEL + e] = f2bf(oacc[jd][r] / lrow[r]);
    }
}

// ---------------------------------------------------------------- launch
extern "C" void kernel_launch(void* const* d_in, const int* in_sizes, int n_in,
                              void* d_out, int out_size, void* d_ws, size_t ws_size,
                              hipStream_t stream) {
  const float* x  = (const float*)d_in[0];
  const float* wq = (const float*)d_in[1];
  const float* bq = (const float*)d_in[2];
  const float* wk = (const float*)d_in[3];
  const float* bk = (const float*)d_in[4];
  const float* wv = (const float*)d_in[5];
  const float* bvv= (const float*)d_in[6];
  const float* wo = (const float*)d_in[7];
  const float* bo = (const float*)d_in[8];

  const size_t NX = (size_t)MTOT * D_MODEL;      // 4M elements
  const size_t NW = (size_t)D_MODEL * D_MODEL;   // 1M elements
  const size_t need = (4 * NX + 4 * NW) * sizeof(unsigned short); // ~40 MB
  if (ws_size < need) return;

  unsigned short* xb  = (unsigned short*)d_ws;
  unsigned short* wqb = xb  + NX;
  unsigned short* wkb = wqb + NW;
  unsigned short* wvb = wkb + NW;
  unsigned short* wob = wvb + NW;
  unsigned short* qb  = wob + NW;
  unsigned short* kb  = qb  + NX;
  unsigned short* vtb = kb  + NX;
  unsigned short* ab  = xb;  // reuse x's bf16 copy (dead after projections)

  const float scale = 0.125f;  // DH^-0.5, folded into Q post-bias

  cvt_f32_bf16<<<dim3((unsigned)((NX + 255) / 256)), 256, 0, stream>>>(x,  xb,  (int)NX);
  cvt_f32_bf16<<<dim3((unsigned)((NW + 255) / 256)), 256, 0, stream>>>(wq, wqb, (int)NW);
  cvt_f32_bf16<<<dim3((unsigned)((NW + 255) / 256)), 256, 0, stream>>>(wk, wkb, (int)NW);
  cvt_f32_bf16<<<dim3((unsigned)((NW + 255) / 256)), 256, 0, stream>>>(wv, wvb, (int)NW);
  cvt_f32_bf16<<<dim3((unsigned)((NW + 255) / 256)), 256, 0, stream>>>(wo, wob, (int)NW);

  dim3 gg(MTOT / 128, D_MODEL / 64);  // (32, 16), 128 threads = 4 waves
  gemm_wmma<<<gg, 128, 0, stream>>>(xb, wqb, bq,  qb,  MTOT, D_MODEL, D_MODEL, scale, 0);
  gemm_wmma<<<gg, 128, 0, stream>>>(xb, wkb, bk,  kb,  MTOT, D_MODEL, D_MODEL, 1.0f,  0);
  gemm_wmma<<<gg, 128, 0, stream>>>(xb, wvb, bvv, vtb, MTOT, D_MODEL, D_MODEL, 1.0f,  1);

  flash_attn<<<dim3(NB * NH, SEQ / 64), 128, 0, stream>>>(qb, kb, vtb, ab);

  gemm_wmma<<<gg, 128, 0, stream>>>(ab, wob, bo, d_out, MTOT, D_MODEL, D_MODEL, 1.0f, 2);
}
